// MultiModalSplitNorm_26491358281733
// MI455X (gfx1250) — matrix operand order repaired and checked by reading, not compiled, not run on hardware
//
#include <hip/hip_runtime.h>
#include <stdint.h>

typedef __attribute__((ext_vector_type(16))) __bf16       v16bf;
typedef __attribute__((ext_vector_type(8)))  float        v8f;
typedef __attribute__((ext_vector_type(4)))  float        f32x4;
typedef __attribute__((ext_vector_type(4)))  unsigned int u32x4;

#define HIDDEN   4096
#define GROUPSZ  4096
#define NMOD     4
#define BM       128            // block tile M
#define BN       256            // block tile N
#define KT       64             // K per LDS stage = 2 WMMA k-steps
#define NSTAGES  (HIDDEN / KT)  // 64
#define LDSB     72             // padded LDS row stride (bf16): 144B = 16B-aligned, bank-conflict-free

union Frag16 { u32x4 q[2]; v16bf v; };

// fp32 -> bf16 RNE, two at a time, packed into one dword
#if __has_builtin(__builtin_amdgcn_cvt_pk_bf16_f32)
__device__ __forceinline__ unsigned pack2_bf16(float a, float b) {
  auto r = __builtin_amdgcn_cvt_pk_bf16_f32(a, b);   // single VALU op
  return __builtin_bit_cast(unsigned, r);
}
#else
__device__ __forceinline__ unsigned pack2_bf16(float a, float b) {
  unsigned ua = __builtin_bit_cast(unsigned, a);
  unsigned ub = __builtin_bit_cast(unsigned, b);
  ua += 0x7FFFu + ((ua >> 16) & 1u);
  ub += 0x7FFFu + ((ub >> 16) & 1u);
  return (ua >> 16) | (ub & 0xFFFF0000u);
}
#endif

// ---- B tile (256x64 bf16): async DMA global -> LDS, ASYNCcnt-tracked ----
__device__ __forceinline__ void issue_async_B(const unsigned short* __restrict__ Wg,
                                              unsigned short* Bbuf, int k0, int t) {
#pragma unroll
  for (int i = 0; i < 8; ++i) {
    int s   = t + i * 256;
    int row = s >> 3;           // 256 rows x 8 segs of 8 bf16
    int cs  = (s & 7) * 8;
    unsigned ldsa = (unsigned)(uintptr_t)(Bbuf + row * LDSB + cs);   // low32 of flat = LDS offset
    uint64_t ga   = (uint64_t)(uintptr_t)(Wg + (size_t)row * HIDDEN + k0 + cs);
    asm volatile("global_load_async_to_lds_b128 %0, %1, off"
                 :: "v"(ldsa), "v"(ga) : "memory");
  }
}

// ---- A tile (128x64 fp32): global -> regs (issued early) ----
__device__ __forceinline__ void load_A_regs(const float* __restrict__ Xg, int k0, int t,
                                            f32x4* areg) {
#pragma unroll
  for (int i = 0; i < 4; ++i) {
    int s   = t + i * 256;
    int row = s >> 3;
    int cs  = (s & 7) * 8;
    const f32x4* src = (const f32x4*)(Xg + (size_t)row * HIDDEN + k0 + cs);
    areg[2 * i]     = src[0];
    areg[2 * i + 1] = src[1];
  }
}

// ---- pack bf16 -> LDS (after compute; co-executes with XDL) ----
__device__ __forceinline__ void store_A_lds(unsigned short* Abuf, int t, const f32x4* areg) {
#pragma unroll
  for (int i = 0; i < 4; ++i) {
    int s   = t + i * 256;
    int row = s >> 3;
    int cs  = (s & 7) * 8;
    u32x4 p;
    p.x = pack2_bf16(areg[2 * i].x,     areg[2 * i].y);
    p.y = pack2_bf16(areg[2 * i].z,     areg[2 * i].w);
    p.z = pack2_bf16(areg[2 * i + 1].x, areg[2 * i + 1].y);
    p.w = pack2_bf16(areg[2 * i + 1].z, areg[2 * i + 1].w);
    *(u32x4*)(Abuf + row * LDSB + cs) = p;   // ds_store_b128
  }
}

// ---- 2 k-steps x (4x4) WMMA from one LDS stage: 1.0 ds_load_b128 per WMMA ----
__device__ __forceinline__ void compute_stage(const unsigned short* Ab,
                                              const unsigned short* Bb,
                                              const int* aBase, const int* bBase,
                                              v8f (&acc)[4][4]) {
#pragma unroll
  for (int ks = 0; ks < 2; ++ks) {
    Frag16 b[4];
#pragma unroll
    for (int nt = 0; nt < 4; ++nt) {
      int o = bBase[nt] + ks * 32;
      b[nt].q[0] = *(const u32x4*)(Bb + o);
      b[nt].q[1] = *(const u32x4*)(Bb + o + 8);
    }
#pragma unroll
    for (int mt = 0; mt < 4; ++mt) {
      Frag16 a;
      int o = aBase[mt] + ks * 32;
      a.q[0] = *(const u32x4*)(Ab + o);
      a.q[1] = *(const u32x4*)(Ab + o + 8);
#pragma unroll
      for (int nt = 0; nt < 4; ++nt)      // 4 back-to-back WMMAs per A fragment
        acc[mt][nt] = __builtin_amdgcn_wmma_f32_16x16x32_bf16(
            false, a.v, false, b[nt].v, (short)0, acc[mt][nt], false, false);
    }
  }
}

__global__ __launch_bounds__(256)
void grouped_linear_bf16_wmma(const float* __restrict__ X,
                              const unsigned short* __restrict__ W,
                              float* __restrict__ Y)
{
  __shared__ __align__(16) unsigned short AsU[2][BM * LDSB];  // 36,864 B
  __shared__ __align__(16) unsigned short BsU[2][BN * LDSB];  // 73,728 B

  const int t    = threadIdx.x;
  const int lane = t & 31;
  const int wave = t >> 5;
  const int wm   = wave >> 2;   // 0..1 (M), 64 rows each
  const int wn   = wave & 3;    // 0..3 (N), 64 cols each
  const int l15  = lane & 15;
  const int hi   = lane >> 4;

  const int g     = blockIdx.z;
  const int mBase = blockIdx.y * BM;
  const int nBase = blockIdx.x * BN;

  const float*          Xg = X + ((size_t)(g * GROUPSZ + mBase)) * HIDDEN;
  const unsigned short* Wg = W + ((size_t)g * HIDDEN + nBase) * HIDDEN;
  float*                Yg = Y + ((size_t)(g * GROUPSZ + mBase)) * HIDDEN + nBase;

  // Per-lane fragment base offsets (bf16 elems) within one stage buffer.
  int aBase[4], bBase[4];
#pragma unroll
  for (int mt = 0; mt < 4; ++mt)
    aBase[mt] = (wm * 64 + mt * 16 + l15) * LDSB + hi * 16;
#pragma unroll
  for (int nt = 0; nt < 4; ++nt)
    bBase[nt] = (wn * 64 + nt * 16 + l15) * LDSB + hi * 16;

  const v8f vzero = {};
  v8f acc[4][4];
#pragma unroll
  for (int mt = 0; mt < 4; ++mt)
#pragma unroll
    for (int nt = 0; nt < 4; ++nt)
      acc[mt][nt] = vzero;

  // ---- Prologue: stage 0 into buffer 0 ----
  issue_async_B(Wg, BsU[0], 0, t);
  {
    f32x4 areg[8];
    load_A_regs(Xg, 0, t, areg);
    store_A_lds(AsU[0], t, areg);
  }
  asm volatile("s_wait_asynccnt 0x0" ::: "memory");
  __syncthreads();

  // ---- Pipelined main loop: one barrier per 64-deep K stage ----
  for (int st = 0; st < NSTAGES; ++st) {
    const int cur = st & 1;
    const int nxt = cur ^ 1;
    const bool more = (st + 1) < NSTAGES;
    const int k1 = (st + 1) * KT;

    f32x4 areg[8];
    if (more) {
      issue_async_B(Wg, BsU[nxt], k1, t);   // B: DMA straight into next buffer
      load_A_regs(Xg, k1, t, areg);         // A: loads in flight during compute
    }

    compute_stage(AsU[cur], BsU[cur], aBase, bBase, acc);   // 32 WMMAs / wave

    if (more)
      store_A_lds(AsU[nxt], t, areg);       // convert + park A into next buffer

    asm volatile("s_wait_asynccnt 0x0" ::: "memory");       // B DMA committed to LDS
    __syncthreads();
  }

  // ---- Store C (fp32, streamed once -> nontemporal) ----
#pragma unroll
  for (int mt = 0; mt < 4; ++mt) {
#pragma unroll
    for (int nt = 0; nt < 4; ++nt) {
      const int mRow0 = wm * 64 + mt * 16 + hi * 8;
      const int nCol  = wn * 64 + nt * 16 + l15;
      float* dst = Yg + (size_t)mRow0 * HIDDEN + nCol;
#pragma unroll
      for (int r = 0; r < 8; ++r)
        __builtin_nontemporal_store(acc[mt][nt][r], dst + (size_t)r * HIDDEN);
    }
  }
}

extern "C" void kernel_launch(void* const* d_in, const int* in_sizes, int n_in,
                              void* d_out, int out_size, void* d_ws, size_t ws_size,
                              hipStream_t stream) {
  (void)in_sizes; (void)n_in; (void)out_size; (void)d_ws; (void)ws_size;
  const float*          X = (const float*)d_in[0];
  const unsigned short* W = (const unsigned short*)d_in[1]; // bf16 raw bits
  float* Y = (float*)d_out;

  dim3 grid(HIDDEN / BN, GROUPSZ / BM, NMOD);  // 16 x 32 x 4
  dim3 block(256);                             // 8 wave32, 2x4 waves of 64x64
  grouped_linear_bf16_wmma<<<grid, block, 0, stream>>>(X, W, Y);
}
